// PairformerLayer_18373870092377
// MI455X (gfx1250) — matrix operand order repaired and checked by reading, not compiled, not run on hardware
//
#include <hip/hip_runtime.h>
#include <hip/hip_bf16.h>
#include <stdint.h>

// ============================================================================
// Pairformer layer for MI455X (gfx1250, wave32, WMMA).
// Heavy GEMMs: v_wmma_f32_16x16x32_bf16, fp32 accumulate, LDS double-buffered,
// incremental staging addressing (uniform-stride pointer walks).
// Workspace requirement: ~1.12 GB (see layout in kernel_launch).
// Input order assumed = setup_inputs() dict insertion order, recursively.
// ============================================================================

typedef __attribute__((ext_vector_type(16))) __bf16 bf16x16;
typedef __attribute__((ext_vector_type(8)))  float  f32x8;

__device__ __forceinline__ uint32_t f2bf_u(float f) {
  union { float f; uint32_t u; } c; c.f = f;
  return (c.u + 0x7FFFu + ((c.u >> 16) & 1u)) >> 16;  // RNE
}
__device__ __forceinline__ uint32_t pack2bf(float lo, float hi) {
  return f2bf_u(lo) | (f2bf_u(hi) << 16);
}
__device__ __forceinline__ float sigmoidf(float x) { return 1.f / (1.f + __expf(-x)); }

// ---------------------------------------------------------------------------
// Generic strided batched GEMM:  C = epilogue( alpha * A x B )
//   A(m,k,chunk,z) = A + m*RAm + chunk*AkC + k*RAk + z*Az      (fp32 -> bf16)
//   B(n,k,chunk,z) = B + n*RBn + chunk*BkC + k*RBk + z*Bz      (fp32 -> bf16)
//   C(m,n,z)       = C + m*RCm + n*RCn + z*Cz
//   Cin(m,n,z)     = I + m*RIm + n*RIn + z*Iz
// mode 0: C = alpha*AB + beta*Cin + bias[n]
// mode 1: C = sigmoid(Cin) * alpha*AB
// mode 2: C = silu(Cin)    * alpha*AB
// ---------------------------------------------------------------------------
struct GemmP {
  const float* A; long long RAm, RAk, AkC, Az;
  const float* B; long long RBn, RBk, BkC, Bz;
  float*       C; long long RCm, RCn, Cz;
  const float* I; long long RIm, RIn, Iz;
  const float* bias;
  float alpha, beta;
  int M, N, Kc, mode;
};

#define BM 128
#define BN 128
#define LDP 40   // padded LDS row stride in ushort (80B, 16B-aligned)

template <bool GUARD>
__launch_bounds__(256)
__global__ void gemm_wmma_bf16(GemmP p) {
  __shared__ unsigned short As[2][BM * LDP];   // A tile [m][k], bf16
  __shared__ unsigned short Bs[2][BN * LDP];   // B tile [n][k], bf16 (k-major)

  const int tid  = threadIdx.x;
  const int wave = tid >> 5, lane = tid & 31;
  const int wm   = wave & 3, wn = wave >> 2;     // waves in 4(m) x 2(n)
  const int l15  = lane & 15, lh = lane >> 4;
  const int bm0  = blockIdx.y * BM, bn0 = blockIdx.x * BN;
  const long long z = blockIdx.z;

  const float* Ab = p.A + z * p.Az;
  const float* Bb = p.B + z * p.Bz;

  // ---- per-thread staging geometry: affine in e, strides uniform ----
  const int am0  = tid >> 4;          // A row at e=0 (advances 16 per e)
  const int akp  = tid & 15;          // A k-pair (constant across e)
  const int bn   = tid & 127;         // B col (constant across e)
  const int bkp0 = tid >> 7;          // B k-pair at e=0 (advances 2 per e)

  const long long RAk = p.RAk, RBk = p.RBk;
  const long long aEstep = 16 * p.RAm;   // uniform per-e stride (A)
  const long long bEstep = 4 * RBk;      // uniform per-e stride (B)

  // base pointers at (e=0, kc=0)
  const float* aP0 = Ab + (long long)(bm0 + am0) * p.RAm + (long long)(2 * akp) * RAk;
  const float* bP0;
  bool bOK = true;
  long long aCl[8]; bool aOK[8];
  if (GUARD) {
    long long gn = bn0 + bn;
    bOK = gn < p.N;
    long long gnc = bOK ? gn : (long long)(p.N - 1);
    bP0 = Bb + gnc * p.RBn + (long long)(2 * bkp0) * RBk;
#pragma unroll
    for (int e = 0; e < 8; ++e) {
      long long gm = bm0 + am0 + 16 * e;
      aOK[e] = gm < p.M;
      long long gmc = aOK[e] ? gm : (long long)(p.M - 1);
      aCl[e] = gmc * p.RAm + (long long)(2 * akp) * RAk;
    }
  } else {
    bP0 = Bb + (long long)(bn0 + bn) * p.RBn + (long long)(2 * bkp0) * RBk;
  }

  // branch-free staging: two consecutive-k floats -> packed u32 -> ds_store_b32
  auto stage = [&](int kc, int buf) {
    unsigned short* asBase = &As[buf][am0 * LDP + 2 * akp];
    unsigned short* bsBase = &Bs[buf][bn * LDP + 2 * bkp0];
    if (GUARD) {
      const long long ck = (long long)kc * p.AkC;
#pragma unroll
      for (int e = 0; e < 8; ++e) {
        float v0 = Ab[aCl[e] + ck];
        float v1 = Ab[aCl[e] + ck + RAk];
        if (!aOK[e]) { v0 = 0.f; v1 = 0.f; }
        *(uint32_t*)(asBase + e * (16 * LDP)) = pack2bf(v0, v1);
      }
      const float* bp = bP0 + (long long)kc * p.BkC;
#pragma unroll
      for (int e = 0; e < 8; ++e) {
        float v0 = bp[0], v1 = bp[RBk];
        if (!bOK) { v0 = 0.f; v1 = 0.f; }
        *(uint32_t*)(bsBase + e * 4) = pack2bf(v0, v1);
        bp += bEstep;
      }
    } else {
      const float* ap = aP0 + (long long)kc * p.AkC;
#pragma unroll
      for (int e = 0; e < 8; ++e) {
        *(uint32_t*)(asBase + e * (16 * LDP)) = pack2bf(ap[0], ap[RAk]);
        ap += aEstep;
      }
      const float* bp = bP0 + (long long)kc * p.BkC;
#pragma unroll
      for (int e = 0; e < 8; ++e) {
        *(uint32_t*)(bsBase + e * 4) = pack2bf(bp[0], bp[RBk]);
        bp += bEstep;
      }
    }
  };

  f32x8 acc[2][4];
#pragma unroll
  for (int i = 0; i < 2; ++i)
#pragma unroll
    for (int j = 0; j < 4; ++j)
#pragma unroll
      for (int r = 0; r < 8; ++r) acc[i][j][r] = 0.f;

  stage(0, 0);
  __syncthreads();

  int buf = 0;
  for (int kc = 0; kc < p.Kc; ++kc) {
    if (kc + 2 < p.Kc) {  // speculative prefetch two chunks ahead -> global_prefetch_b8
      __builtin_prefetch(aP0 + (long long)(kc + 2) * p.AkC, 0, 1);
      __builtin_prefetch(bP0 + (long long)(kc + 2) * p.BkC, 0, 1);
    }
    if (kc + 1 < p.Kc)    // stage next chunk into the other buffer (overlaps WMMA)
      stage(kc + 1, buf ^ 1);

    // ---- fragments per documented 16-bit layouts ----
    union Fr { bf16x16 v; uint4 q[2]; };
    Fr af[2], bfx[4];
#pragma unroll
    for (int sm = 0; sm < 2; ++sm) {
      int m = wm * 32 + sm * 16 + l15;
      const uint4* rp = reinterpret_cast<const uint4*>(&As[buf][m * LDP]);
      af[sm].q[0] = rp[lh];       // K = 8*half + 0..7
      af[sm].q[1] = rp[2 + lh];   // K = 16 + 8*half + 0..7
    }
#pragma unroll
    for (int sn = 0; sn < 4; ++sn) {
      int n = wn * 64 + sn * 16 + l15;
      const uint4* rp = reinterpret_cast<const uint4*>(&Bs[buf][n * LDP]);
      bfx[sn].q[0] = rp[lh];          // K = 8*half + 0..7
      bfx[sn].q[1] = rp[2 + lh];      // K = 16 + 8*half + 0..7
    }
#pragma unroll
    for (int sm = 0; sm < 2; ++sm)
#pragma unroll
      for (int sn = 0; sn < 4; ++sn)
        acc[sm][sn] = __builtin_amdgcn_wmma_f32_16x16x32_bf16(
            false, af[sm].v, false, bfx[sn].v, (short)0, acc[sm][sn], false, false);

    __syncthreads();
    buf ^= 1;
  }

  // ---- epilogue / store (C/D layout: lane n=l15, vgpr r -> m = r + 8*half) ----
  float*       Cb = p.C + z * p.Cz;
  const float* Ib = p.I ? (p.I + z * p.Iz) : nullptr;
#pragma unroll
  for (int sm = 0; sm < 2; ++sm)
#pragma unroll
    for (int sn = 0; sn < 4; ++sn)
#pragma unroll
      for (int r = 0; r < 8; ++r) {
        int gm = bm0 + wm * 32 + sm * 16 + r + 8 * lh;
        int gn = bn0 + wn * 64 + sn * 16 + l15;
        if (!GUARD || (gm < p.M && gn < p.N)) {
          float v = p.alpha * acc[sm][sn][r];
          if (p.mode == 0) {
            if (Ib) v += p.beta * Ib[(long long)gm * p.RIm + (long long)gn * p.RIn];
            if (p.bias) v += p.bias[gn];
          } else {
            float ci = Ib[(long long)gm * p.RIm + (long long)gn * p.RIn];
            float s  = sigmoidf(ci);
            v *= (p.mode == 1) ? s : ci * s;
          }
          Cb[(long long)gm * p.RCm + (long long)gn * p.RCn] = v;
        }
      }
}

// ---------------------------------------------------------------------------
// LayerNorm over last dim D, one 128-thread block per row.
// ---------------------------------------------------------------------------
__global__ void ln_kernel(const float* __restrict__ x, const float* __restrict__ g,
                          const float* __restrict__ b, float* __restrict__ out, int D) {
  long long row = blockIdx.x;
  const float* xr = x + row * (long long)D;
  float* orow = out + row * (long long)D;
  int t = threadIdx.x;
  float s1 = 0.f, s2 = 0.f;
  for (int i = t; i < D; i += 128) { float v = xr[i]; s1 += v; s2 += v * v; }
  __shared__ float r1[128], r2[128];
  r1[t] = s1; r2[t] = s2; __syncthreads();
  for (int o = 64; o > 0; o >>= 1) {
    if (t < o) { r1[t] += r1[t + o]; r2[t] += r2[t + o]; }
    __syncthreads();
  }
  float mu = r1[0] / D;
  float var = r2[0] / D - mu * mu;
  float rs = rsqrtf(var + 1e-5f);
  for (int i = t; i < D; i += 128) orow[i] = (xr[i] - mu) * rs * g[i] + b[i];
}

// a = sigmoid(p0)*p1 ; b = sigmoid(p2)*p3 from the 4*TC projection
__global__ void tri_gate_kernel(const float* __restrict__ proj, float* __restrict__ a,
                                float* __restrict__ b, long long n) {
  long long i = blockIdx.x * 256LL + threadIdx.x;
  if (i >= n) return;
  long long r = i >> 7; int c = (int)(i & 127);
  const float* pr = proj + r * 512;
  a[i] = sigmoidf(pr[c])       * pr[128 + c];
  b[i] = sigmoidf(pr[256 + c]) * pr[384 + c];
}

__global__ void add_kernel(float* __restrict__ dst, const float* __restrict__ src, long long n) {
  long long i = blockIdx.x * 256LL + threadIdx.x;
  if (i < n) dst[i] += src[i];
}

// row softmax, 256 threads per row
__global__ void softmax_kernel(float* __restrict__ x, int L) {
  long long row = blockIdx.x;
  float* xr = x + row * (long long)L;
  __shared__ float red[256];
  int t = threadIdx.x;
  float m = -3.4e38f;
  for (int i = t; i < L; i += 256) m = fmaxf(m, xr[i]);
  red[t] = m; __syncthreads();
  for (int o = 128; o > 0; o >>= 1) { if (t < o) red[t] = fmaxf(red[t], red[t + o]); __syncthreads(); }
  m = red[0]; __syncthreads();
  float s = 0.f;
  for (int i = t; i < L; i += 256) { float e = __expf(xr[i] - m); xr[i] = e; s += e; }
  red[t] = s; __syncthreads();
  for (int o = 128; o > 0; o >>= 1) { if (t < o) red[t] += red[t + o]; __syncthreads(); }
  float inv = 1.f / red[0];
  for (int i = t; i < L; i += 256) xr[i] *= inv;
}

// Vsum[h][c] = sum over (i,j) of v = proj[h,i,j,64+c]   (faithful size-1 broadcast)
__global__ void vsum_kernel(const float* __restrict__ proj, float* __restrict__ vs) {
  int h = blockIdx.x >> 5, c = blockIdx.x & 31;
  const float* base = proj + (long long)h * 33554432LL + 64 + c;
  float s = 0.f;
  for (long long r = threadIdx.x; r < 262144LL; r += 256) s += base[r * 128];
  __shared__ float red[256];
  red[threadIdx.x] = s; __syncthreads();
  for (int o = 128; o > 0; o >>= 1) { if (threadIdx.x < o) red[threadIdx.x] += red[threadIdx.x + o]; __syncthreads(); }
  if (threadIdx.x == 0) vs[blockIdx.x] = red[0];
}

// outf[h,i,j,c](torch-flat) = g[h,i,j,c] * softmax[h,i,j] * Vsum[h,c]
__global__ void tri_outf_kernel(const float* __restrict__ proj, const float* __restrict__ sm,
                                const float* __restrict__ vs, float* __restrict__ outf) {
  long long idx = blockIdx.x * 256LL + threadIdx.x;
  if (idx >= 33554432LL) return;
  int c = (int)(idx & 31);
  long long tt = idx >> 5;
  int j = (int)(tt & 511); tt >>= 9;
  int i = (int)(tt & 511); int h = (int)(tt >> 9);
  float g = proj[(long long)h * 33554432LL + (long long)i * 65536 + j * 128 + 96 + c];
  float a = sm[(long long)h * 262144 + i * 512 + j];
  outf[(long long)h * 8388608LL + (long long)i * 16384 + j * 32 + c] = g * a * vs[h * 32 + c];
}

// pair attention: score[h,j,i] = (q[h,i,:].k[h,j,:])/24 + bias_flat[h,j,i]
__global__ void pair_score_kernel(const float* __restrict__ q, const float* __restrict__ kvg,
                                  const float* __restrict__ biasf, float* __restrict__ score) {
  long long idx = blockIdx.x * 256LL + threadIdx.x;
  if (idx >= 4194304LL) return;
  int i = (int)(idx & 511);
  int j = (int)((idx >> 9) & 511);
  int h = (int)(idx >> 18);
  const float* qr = q + (long long)h * 12288 + i * 24;
  const float* kr = kvg + (long long)h * 36864 + j * 72;
  float s = 0.f;
#pragma unroll
  for (int c = 0; c < 24; ++c) s += qr[c] * kr[c];
  score[idx] = s * (1.f / 24.f) + biasf[idx];
}

// out[h,j,c] = sigmoid(g[h,j,c]) * sum_i A[h,i,j]*v[h,i,c]
__global__ void pair_out_kernel(const float* __restrict__ sm, const float* __restrict__ kvg,
                                float* __restrict__ out2) {
  int idx = blockIdx.x * 256 + threadIdx.x;
  if (idx >= 196608) return;
  int c = idx % 24;
  int j = (idx / 24) % 512;
  int h = idx / 12288;
  const float* smh = sm + (long long)h * 262144 + j;      // over i, stride 512
  const float* vh  = kvg + (long long)h * 36864 + 24 + c; // over i, stride 72
  float acc = 0.f;
  for (int i = 0; i < 512; ++i) acc += smh[(long long)i * 512] * vh[(long long)i * 72];
  float g = sigmoidf(kvg[(long long)h * 36864 + (long long)j * 72 + 48 + c]);
  out2[idx] = g * acc;
}

// ---------------------------------------------------------------------------
// Host-side launch helpers
// ---------------------------------------------------------------------------
static void gemm_full(hipStream_t st,
    const float* A, long long RAm, long long RAk, long long AkC, long long Az,
    const float* B, long long RBn, long long RBk, long long BkC, long long Bz,
    float* C, long long RCm, long long RCn, long long Cz,
    const float* I, long long RIm, long long RIn, long long Iz,
    const float* bias, float alpha, float beta,
    int M, int N, int Kc, int mode, int batches) {
  GemmP p{A, RAm, RAk, AkC, Az, B, RBn, RBk, BkC, Bz, C, RCm, RCn, Cz,
          I, RIm, RIn, Iz, bias, alpha, beta, M, N, Kc, mode};
  dim3 grid((unsigned)((N + 127) / 128), (unsigned)((M + 127) / 128), (unsigned)batches);
  if ((M & 127) || (N & 127))
    gemm_wmma_bf16<true><<<grid, dim3(256), 0, st>>>(p);
  else
    gemm_wmma_bf16<false><<<grid, dim3(256), 0, st>>>(p);
}

// X (M x K, row-major, lda=K)  @  W (row stride ldw, first col at W)  -> C (ldc)
static void gemm_xw(hipStream_t st, const float* X, int K, const float* W, long long ldw,
                    float* C, long long ldc, int M, int N,
                    const float* bias, const float* I, long long ldi, float beta, int mode,
                    float alpha = 1.f) {
  gemm_full(st, X, K, 1, 32, 0,
            W, 1, ldw, 32 * ldw, 0,
            C, ldc, 1, 0,
            I, ldi, 1, 0,
            bias, alpha, beta, M, N, K / 32, mode, 1);
}

static inline unsigned gdiv(long long n) { return (unsigned)((n + 255) / 256); }

extern "C" void kernel_launch(void* const* d_in, const int* in_sizes, int n_in,
                              void* d_out, int out_size, void* d_ws, size_t ws_size,
                              hipStream_t stream) {
  (void)in_sizes; (void)n_in; (void)out_size; (void)ws_size;
  const long long N2 = 262144;           // 512*512
  const long long ZN = N2 * 128;         // z elements
  const long long SN = 512 * 384;        // s elements

  // ---- inputs (setup_inputs insertion order, recursively flattened) ----
  const float* s_in = (const float*)d_in[0];
  const float* z_in = (const float*)d_in[1];
  auto P = [&](int i) { return (const float*)d_in[i]; };
  // tmo:2..8  tmi:9..15  tas:16..20  tae:21..25  trans_z:26..30  trans_s:31..35  attn:36..45

  // ---- outputs ----
  float* sres = (float*)d_out;       // 512x384
  float* zres = sres + SN;           // 512x512x128

  // ---- workspace layout (floats); total ~279M floats (~1.12 GB) ----
  float* W0 = (float*)d_ws;
  float* S0  = W0;                 // N2*512  proj / transition hidden
  float* S1  = S0 + N2 * 512;      // N2*128
  float* S2  = S1 + ZN;            // N2*128
  float* S3  = S2 + ZN;            // N2*128
  float* S4  = S3 + ZN;            // N2*128
  float* SB  = S4 + ZN;            // N2*16  (bias projections)
  float* SC  = SB + N2 * 16;       // N2*16  (scores: up to 16*512*512)
  float* VS  = SC + N2 * 16;       // 128
  float* ALN = VS + 128;           // 512*384
  float* QB  = ALN + SN;           // 512*384
  float* KVG = QB + SN;            // 512*1152
  float* H1  = KVG + 512 * 1152;   // 512*1536
  float* O2  = H1 + 512 * 1536;    // 512*384

  // ---- init residual accumulators ----
  hipMemcpyAsync(sres, s_in, SN * sizeof(float), hipMemcpyDeviceToDevice, stream);
  hipMemcpyAsync(zres, z_in, ZN * sizeof(float), hipMemcpyDeviceToDevice, stream);

  // ======================= tri_mul (outgoing / incoming) ====================
  for (int pass = 0; pass < 2; ++pass) {
    int o = (pass == 0) ? 2 : 9;                 // tmo / tmi param base
    bool outgoing = (pass == 0);
    ln_kernel<<<dim3((unsigned)N2), dim3(128), 0, stream>>>(zres, P(o + 0), P(o + 1), S1, 128);
    gemm_xw(stream, S1, 128, P(o + 2), 512, S0, 512, (int)N2, 512, nullptr, nullptr, 0, 0.f, 0);
    tri_gate_kernel<<<gdiv(ZN), 256, 0, stream>>>(S0, S2, S3, ZN);
    gemm_xw(stream, S1, 128, P(o + 3), 128, S4, 128, (int)N2, 128, nullptr, nullptr, 0, 0.f, 0);
    if (outgoing) {
      // ab[i,j,d] = sum_k a[i,k,d] * b[j,k,d]   (batched over d=128)
      gemm_full(stream, S2, 65536, 128, 4096, 1,
                        S3, 65536, 128, 4096, 1,
                        S1, 65536, 128, 1,
                        nullptr, 0, 0, 0, nullptr, 1.f, 0.f,
                512, 512, 16, 0, 128);
    } else {
      // ab[i,j,d] = sum_k a[k,i,d] * b[k,j,d]
      gemm_full(stream, S2, 128, 65536, 2097152, 1,
                        S3, 128, 65536, 2097152, 1,
                        S1, 65536, 128, 1,
                        nullptr, 0, 0, 0, nullptr, 1.f, 0.f,
                512, 512, 16, 0, 128);
    }
    ln_kernel<<<dim3((unsigned)N2), dim3(128), 0, stream>>>(S1, P(o + 4), P(o + 5), S2, 128);
    // sigmoid(x@gate_w) * (LN(ab)@ab_w)
    gemm_xw(stream, S2, 128, P(o + 6), 128, S3, 128, (int)N2, 128, nullptr, S4, 128, 0.f, 1);
    add_kernel<<<gdiv(ZN), 256, 0, stream>>>(zres, S3, ZN);
  }

  // ======================= tri_attn (starting / ending) =====================
  for (int pass = 0; pass < 2; ++pass) {
    int o = (pass == 0) ? 16 : 21;               // tas / tae
    bool ending = (pass == 1);
    ln_kernel<<<dim3((unsigned)N2), dim3(128), 0, stream>>>(zres, P(o + 0), P(o + 1), S1, 128);
    gemm_xw(stream, S1, 128, P(o + 2), 512, S0, 512, (int)N2, 512, nullptr, nullptr, 0, 0.f, 0);
    gemm_xw(stream, S1, 128, P(o + 3), 4, SB, 4, (int)N2, 4, nullptr, nullptr, 0, 0.f, 0);
    vsum_kernel<<<dim3(128), dim3(256), 0, stream>>>(S0, VS);
    // score[h,j,k'] = (1/sqrt(128)) * sum_{i,c} q[h,i,j,c]*k[h,i,k',c] + bias
    gemm_full(stream,
              S0,        128, 1, 65536, 33554432,        // q: A(m=j, chunk=i, k=c)
              S0 + 32,   128, 1, 65536, 33554432,        // k: B(n=k', chunk=i, k=c)
              SC,        512, 1, 262144,
              SB, ending ? 4 : 2048, ending ? 2048 : 4, 1,  // bias[h,x,y] (or swapped)
              nullptr, rsqrtf(128.f), 1.f,
              512, 512, 512, 0, 4);
    softmax_kernel<<<dim3(2048), dim3(256), 0, stream>>>(SC, 512);
    tri_outf_kernel<<<gdiv(ZN), 256, 0, stream>>>(S0, SC, VS, S2);
    // z += outf @ out_w
    gemm_xw(stream, S2, 128, P(o + 4), 128, zres, 128, (int)N2, 128, nullptr, zres, 128, 1.f, 0);
  }

  // ======================= transition (z) ===================================
  ln_kernel<<<dim3((unsigned)N2), dim3(128), 0, stream>>>(zres, P(26), P(27), S1, 128);
  gemm_xw(stream, S1, 128, P(28),       1024, S0, 512, (int)N2, 512, nullptr, nullptr, 0, 0.f, 0);
  gemm_xw(stream, S1, 128, P(28) + 512, 1024, S0, 512, (int)N2, 512, nullptr, S0, 512, 0.f, 2); // silu(a)*b
  gemm_xw(stream, S0, 512, P(29), 128, zres, 128, (int)N2, 128, P(30), zres, 128, 1.f, 0);

  // ======================= attn_pair_bias (s, z) ============================
  ln_kernel<<<dim3(512), dim3(128), 0, stream>>>(sres, P(36), P(37), ALN, 384);
  gemm_xw(stream, ALN, 384, P(40), 384,  QB,  384,  512, 384,  P(41), nullptr, 0, 0.f, 0);
  gemm_xw(stream, ALN, 384, P(42), 1152, KVG, 1152, 512, 1152, nullptr, nullptr, 0, 0.f, 0);
  ln_kernel<<<dim3((unsigned)N2), dim3(128), 0, stream>>>(zres, P(38), P(39), S1, 128);
  gemm_xw(stream, S1, 128, P(43), 16, SB, 16, (int)N2, 16, P(44), nullptr, 0, 0.f, 0);
  pair_score_kernel<<<gdiv(4194304), 256, 0, stream>>>(QB, KVG, SB, SC);
  softmax_kernel<<<dim3(8192), dim3(256), 0, stream>>>(SC, 512);
  pair_out_kernel<<<gdiv(196608), 256, 0, stream>>>(SC, KVG, O2);
  gemm_xw(stream, O2, 384, P(45), 384, sres, 384, 512, 384, nullptr, sres, 384, 1.f, 0);

  // ======================= transition (s) ===================================
  ln_kernel<<<dim3(512), dim3(128), 0, stream>>>(sres, P(31), P(32), ALN, 384);
  gemm_xw(stream, ALN, 384, P(33),        3072, H1, 1536, 512, 1536, nullptr, nullptr, 0, 0.f, 0);
  gemm_xw(stream, ALN, 384, P(33) + 1536, 3072, H1, 1536, 512, 1536, nullptr, H1, 1536, 0.f, 2);
  gemm_xw(stream, H1, 1536, P(34), 384, sres, 384, 512, 384, P(35), sres, 384, 1.f, 0);
}
